// GLCRClassifier_84524956385954
// MI455X (gfx1250) — compile-verified, matrix operand order checked by reading, not run on hardware
//
#include <hip/hip_runtime.h>
#include <hip/hip_bf16.h>
#include <math.h>

// ---------------------------------------------------------------------------
// Types for CDNA5 WMMA
// ---------------------------------------------------------------------------
typedef _Float16 h16;
typedef __attribute__((ext_vector_type(16))) _Float16 v16h;
typedef __attribute__((ext_vector_type(8)))  _Float16 v8h;
typedef __attribute__((ext_vector_type(8)))  float    v8f;

#define ACT_NONE 0
#define ACT_RELU 1
#define ACT_SIG  2

#define NB 16384          // batch
#define ND 256            // D
#define NK 16             // mediators
#define NBK (NB * NK)
#define CATS 544          // concat buffer stride (multiple of 32, >= 520)

// ---------------------------------------------------------------------------
// GEMM:  Y[R,N] = act( X[R,:Kd] @ W[Kd,N] + bias[N] ), operands f16 in memory.
//   X: f16, row stride ldx (multiple of 8), K zero-padded to kpad.
//   Wt: f16 transposed+padded weights, Wt[n*kpad + k], npad rows (zero pad).
//   Each wave computes a 16 x (NT*16) tile: A loaded once per K-slab,
//   NT WMMAs against NT B streams. Each operand tile is exactly two b128
//   loads (8 halves each) concatenated by shufflevector -> no conversions,
//   minimal register pressure, half the traffic of an f32 operand path.
// Requirements: R % 16 == 0, kpad % 32 == 0, zero pad in X cols [Kd,kpad),
//               (N/16) % NT == 0 when NT > 1.
// ---------------------------------------------------------------------------
template<int ACT, int NT>
__global__ void k_gemm(const h16* __restrict__ X, int ldx,
                       const h16* __restrict__ Wt, int kpad,
                       const float* __restrict__ bias,
                       float* __restrict__ Y, h16* __restrict__ Y16,
                       int ldy, int R, int N)
{
    const int  lane = threadIdx.x & 31;
    const long wave = ((long)blockIdx.x * (blockDim.x >> 5)) + (threadIdx.x >> 5);
    const int  cgroups = ((N + 15) >> 4) / NT;
    const long rtiles  = (long)R >> 4;
    if (wave >= rtiles * cgroups) return;

    const int  cg   = (int)(wave % cgroups);
    const long rt   = wave / cgroups;
    const int  row0 = (int)(rt << 4);
    const int  col0 = cg * (NT * 16);
    const int  half = lane >> 4;
    const int  l15  = lane & 15;

    const h16* __restrict__ xrow = X + (long)(row0 + l15) * ldx + half * 8;
    const h16* wrow[NT];
#pragma unroll
    for (int j = 0; j < NT; ++j)
        wrow[j] = Wt + (long)(col0 + j * 16 + l15) * kpad + half * 8;

    const v8f zero = {0.f, 0.f, 0.f, 0.f, 0.f, 0.f, 0.f, 0.f};
    v8f acc[NT];
#pragma unroll
    for (int j = 0; j < NT; ++j) acc[j] = zero;

    for (int kb = 0; kb < kpad; kb += 32) {
        __builtin_prefetch(xrow + kb + 32, 0, 1);   // speculative next-slab prefetch
        const v8h a0 = *(const v8h*)(xrow + kb);
        const v8h a1 = *(const v8h*)(xrow + kb + 16);
        const v16h a = __builtin_shufflevector(a0, a1,
            0, 1, 2, 3, 4, 5, 6, 7, 8, 9, 10, 11, 12, 13, 14, 15);
#pragma unroll
        for (int j = 0; j < NT; ++j) {
            const v8h b0 = *(const v8h*)(wrow[j] + kb);
            const v8h b1 = *(const v8h*)(wrow[j] + kb + 16);
            const v16h b = __builtin_shufflevector(b0, b1,
                0, 1, 2, 3, 4, 5, 6, 7, 8, 9, 10, 11, 12, 13, 14, 15);
            acc[j] = __builtin_amdgcn_wmma_f32_16x16x32_f16(
                /*neg_a=*/false, a, /*neg_b=*/false, b,
                /*c_mod=*/(short)0, acc[j], /*reuse_a=*/false, /*reuse_b=*/false);
        }
    }

#pragma unroll
    for (int j = 0; j < NT; ++j) {
        const int bcol = col0 + j * 16 + l15;
        if (bcol < N) {
            const float bv = bias ? bias[bcol] : 0.f;
#pragma unroll
            for (int r = 0; r < 8; ++r) {
                const int row = row0 + half * 8 + r;
                float v = acc[j][r] + bv;
                if (ACT == ACT_RELU)     v = fmaxf(v, 0.f);
                else if (ACT == ACT_SIG) v = __builtin_amdgcn_rcpf(1.f + __expf(-v));
                if (Y)   Y[(long)row * ldy + bcol]   = v;
                if (Y16) Y16[(long)row * ldy + bcol] = (h16)v;
            }
        }
    }
}

static void launch_gemm(const h16* X, int ldx, const h16* Wt, int kpad,
                        const float* bias, float* Y, h16* Y16, int ldy,
                        int R, int N, int act, hipStream_t stream)
{
    const int ctiles = (N + 15) / 16;
    const bool big = (N % 64 == 0);
    const int nt = big ? 4 : 1;
    const long tiles = (long)(R / 16) * (ctiles / nt);
    const dim3 grid((unsigned)((tiles + 7) / 8)), block(256);
    if (big) {
        if (act == ACT_NONE)
            k_gemm<ACT_NONE, 4><<<grid, block, 0, stream>>>(X, ldx, Wt, kpad, bias, Y, Y16, ldy, R, N);
        else if (act == ACT_RELU)
            k_gemm<ACT_RELU, 4><<<grid, block, 0, stream>>>(X, ldx, Wt, kpad, bias, Y, Y16, ldy, R, N);
        else
            k_gemm<ACT_SIG, 4><<<grid, block, 0, stream>>>(X, ldx, Wt, kpad, bias, Y, Y16, ldy, R, N);
    } else {
        if (act == ACT_NONE)
            k_gemm<ACT_NONE, 1><<<grid, block, 0, stream>>>(X, ldx, Wt, kpad, bias, Y, Y16, ldy, R, N);
        else if (act == ACT_RELU)
            k_gemm<ACT_RELU, 1><<<grid, block, 0, stream>>>(X, ldx, Wt, kpad, bias, Y, Y16, ldy, R, N);
        else
            k_gemm<ACT_SIG, 1><<<grid, block, 0, stream>>>(X, ldx, Wt, kpad, bias, Y, Y16, ldy, R, N);
    }
}

// ---------------------------------------------------------------------------
// Weight transpose + pad to f16: Wt[n*kpad+k] = (n<N && k<Kd) ? W[k*N+n] : 0
// ---------------------------------------------------------------------------
__global__ void k_wtrans16(const float* __restrict__ W, h16* __restrict__ Wt,
                           int Kd, int N, int kpad, int npad)
{
    const long i = (long)blockIdx.x * blockDim.x + threadIdx.x;
    const long total = (long)npad * kpad;
    if (i >= total) return;
    const int n = (int)(i / kpad), k = (int)(i % kpad);
    Wt[i] = (n < N && k < Kd) ? (h16)W[(long)k * N + n] : (h16)0.f;
}

// ---------------------------------------------------------------------------
// Row LayerNorm over Dd valid cols, stride ld; templated in/out element type.
// ---------------------------------------------------------------------------
template<typename TI, typename TO>
__global__ void k_ln(const TI* __restrict__ X, TO* __restrict__ Y,
                     const float* __restrict__ g, const float* __restrict__ b,
                     int ld, int Dd)
{
    __shared__ float red[256];
    const long row = blockIdx.x;
    const TI* x = X + row * (long)ld;
    TO* y = Y + row * (long)ld;
    const int tid = threadIdx.x;

    float s = 0.f;
    for (int c = tid; c < Dd; c += 256) s += (float)x[c];
    red[tid] = s; __syncthreads();
    for (int off = 128; off > 0; off >>= 1) {
        if (tid < off) red[tid] += red[tid + off];
        __syncthreads();
    }
    const float mean = red[0] / (float)Dd;
    __syncthreads();

    float v = 0.f;
    for (int c = tid; c < Dd; c += 256) { const float d = (float)x[c] - mean; v += d * d; }
    red[tid] = v; __syncthreads();
    for (int off = 128; off > 0; off >>= 1) {
        if (tid < off) red[tid] += red[tid + off];
        __syncthreads();
    }
    const float rstd = rsqrtf(red[0] / (float)Dd + 1e-5f);
    __syncthreads();

    for (int c = tid; c < Dd; c += 256)
        y[c] = (TO)(((float)x[c] - mean) * rstd * g[c] + b[c]);
}

// ---------------------------------------------------------------------------
// Small helpers
// ---------------------------------------------------------------------------
__global__ void k_copyh(h16* __restrict__ dst, int dstride, int dcol,
                        const float* __restrict__ src, int sstride, int scol,
                        long rows, int cols)
{
    const long i = (long)blockIdx.x * blockDim.x + threadIdx.x;
    if (i >= rows * (long)cols) return;
    const long r = i / cols; const int c = (int)(i % cols);
    dst[r * (long)dstride + dcol + c] = (h16)src[r * (long)sstride + scol + c];
}

__global__ void k_cvt16(const float* __restrict__ s, h16* __restrict__ d, long n)
{   const long i = (long)blockIdx.x * blockDim.x + threadIdx.x;
    if (i < n) d[i] = (h16)s[i]; }

__global__ void k_fill0(float* __restrict__ p, long n)
{   const long i = (long)blockIdx.x * blockDim.x + threadIdx.x;
    if (i < n) p[i] = 0.f; }

__global__ void k_fillh(h16* __restrict__ p, long n)
{   const long i = (long)blockIdx.x * blockDim.x + threadIdx.x;
    if (i < n) p[i] = (h16)0.f; }

__global__ void k_fillcolsh(h16* __restrict__ dst, int stride, int col0, int cols, long rows)
{   const long i = (long)blockIdx.x * blockDim.x + threadIdx.x;
    if (i >= rows * (long)cols) return;
    const long r = i / cols; const int c = (int)(i % cols);
    dst[r * (long)stride + col0 + c] = (h16)0.f; }

__global__ void k_pairbuild(const float* __restrict__ xi, const float* __restrict__ xj,
                            float* __restrict__ pair, long Bn, int Dd)
{
    const long i = (long)blockIdx.x * blockDim.x + threadIdx.x;
    const long total = Bn * 4L * Dd;
    if (i >= total) return;
    const long bb = i / (4L * Dd);
    const int  c  = (int)(i % (4L * Dd));
    const int  d  = c % Dd, seg = c / Dd;
    const float a = xi[bb * Dd + d], b = xj[bb * Dd + d];
    float v;
    if (seg == 0) v = a; else if (seg == 1) v = b;
    else if (seg == 2) v = fabsf(a - b); else v = a * b;
    pair[i] = v;
}

__global__ void k_tokin(const float* __restrict__ nodeH, const float* __restrict__ q,
                        const float* __restrict__ featH, const float* __restrict__ prior,
                        const float* __restrict__ prior_scale, float* __restrict__ tok,
                        long BK)
{
    const long i = (long)blockIdx.x * blockDim.x + threadIdx.x;
    const long total = BK * 385L;
    if (i >= total) return;
    const long r = i / 385L; const int c = (int)(i % 385L);
    const long b = r / NK;
    float v;
    if      (c < 128) v = nodeH[r * 128 + c];
    else if (c < 256) v = q[b * 128 + (c - 128)];
    else if (c < 384) v = featH[r * 128 + (c - 256)];
    else              v = prior_scale[0] * prior[r];
    tok[r * 416 + c] = v;
}

__global__ void k_softmax16(const float* __restrict__ sc, const unsigned char* __restrict__ mask,
                            float* __restrict__ outp, int ostride, long rows, float scale)
{
    const long b = (long)blockIdx.x * blockDim.x + threadIdx.x;
    if (b >= rows) return;
    float s[16]; float mx = -3.0e38f;
#pragma unroll
    for (int k = 0; k < 16; ++k) {
        float v = sc[b * 16 + k] * scale;
        if (mask && !mask[b * 16 + k]) v = -1e9f;
        s[k] = v; mx = fmaxf(mx, v);
    }
    float sum = 0.f;
#pragma unroll
    for (int k = 0; k < 16; ++k) { s[k] = __expf(s[k] - mx); sum += s[k]; }
    const float inv = __builtin_amdgcn_rcpf(sum);
#pragma unroll
    for (int k = 0; k < 16; ++k) outp[b * ostride + k] = s[k] * inv;
}

__global__ void k_attnpool(const float* __restrict__ attn, const float* __restrict__ vals,
                           float* __restrict__ outp, long Bn)
{
    const long i = (long)blockIdx.x * blockDim.x + threadIdx.x;
    if (i >= Bn * ND) return;
    const long b = i / ND; const int d = (int)(i % ND);
    float s = 0.f;
#pragma unroll
    for (int k = 0; k < NK; ++k)
        s += attn[b * NK + k] * vals[(b * NK + k) * (long)ND + d];
    outp[i] = s;
}

__global__ void k_ctxpool(const float* __restrict__ tsc, const float* __restrict__ tproj,
                          float* __restrict__ outp, long Bn)
{
    const long i = (long)blockIdx.x * blockDim.x + threadIdx.x;
    if (i >= Bn * ND) return;
    const long b = i / ND; const int d = (int)(i % ND);
    const float s0 = tsc[2 * b], s1 = tsc[2 * b + 1];
    const float mx = fmaxf(s0, s1);
    const float e0 = __expf(s0 - mx), e1 = __expf(s1 - mx);
    const float inv = __builtin_amdgcn_rcpf(e0 + e1);
    outp[i] = (e0 * inv) * tproj[(2 * b) * (long)ND + d]
            + (e1 * inv) * tproj[(2 * b + 1) * (long)ND + d];
}

__global__ void k_fma3(const float* __restrict__ a, const float* __restrict__ g,
                       const float* __restrict__ m, float* __restrict__ y, long n)
{   const long i = (long)blockIdx.x * blockDim.x + threadIdx.x;
    if (i < n) y[i] = a[i] + g[i] * m[i]; }

__global__ void k_mix(const float* __restrict__ g, const float* __restrict__ a,
                      const float* __restrict__ b, float* __restrict__ y, long n)
{   const long i = (long)blockIdx.x * blockDim.x + threadIdx.x;
    if (i < n) y[i] = g[i] * a[i] + (1.f - g[i]) * b[i]; }

__global__ void k_avg(const float* __restrict__ a, const float* __restrict__ b,
                      float* __restrict__ y, long n)
{   const long i = (long)blockIdx.x * blockDim.x + threadIdx.x;
    if (i < n) y[i] = 0.5f * (a[i] + b[i]); }

__global__ void k_mul(const float* __restrict__ a, const float* __restrict__ b,
                      float* __restrict__ y, long n)
{   const long i = (long)blockIdx.x * blockDim.x + threadIdx.x;
    if (i < n) y[i] = a[i] * b[i]; }

__global__ void k_addv(const float* __restrict__ a, const float* __restrict__ b,
                       float* __restrict__ y, long n)
{   const long i = (long)blockIdx.x * blockDim.x + threadIdx.x;
    if (i < n) y[i] = a[i] + b[i]; }

__global__ void k_addconf(float* __restrict__ y, const float* __restrict__ conf,
                          const float* __restrict__ res, int Dd, long n)
{   const long i = (long)blockIdx.x * blockDim.x + threadIdx.x;
    if (i < n) y[i] += (1.f - conf[i / Dd]) * res[i]; }

__global__ void k_gap(const float* __restrict__ gl, const float* __restrict__ bl,
                      h16* __restrict__ dst, int stride, int col0, long Bn)
{   const long i = (long)blockIdx.x * blockDim.x + threadIdx.x;
    if (i >= 2 * Bn) return;
    const long b = i >> 1; const int c = (int)(i & 1);
    dst[b * (long)stride + col0 + c] = (h16)fabsf(gl[i] - bl[i]); }

__global__ void k_final(const float* __restrict__ gl, const float* __restrict__ bl,
                        const float* __restrict__ bbl, const float* __restrict__ baselog,
                        const float* __restrict__ bconf, const float* __restrict__ util,
                        const float* __restrict__ rscale, float* __restrict__ outp, long Bn)
{
    const long i = (long)blockIdx.x * blockDim.x + threadIdx.x;
    if (i >= 2 * Bn) return;
    const long b = i >> 1;
    const float gain = 0.5f + 0.5f * util[b];
    const float contrib = bconf[b] * gain * bbl[i] + gain * (bl[i] - bbl[i]);
    outp[i] = gl[i] + contrib + rscale[0] * baselog[i];
}

// ---------------------------------------------------------------------------
// Host orchestration
// ---------------------------------------------------------------------------
extern "C" void kernel_launch(void* const* d_in, const int* in_sizes, int n_in,
                              void* d_out, int out_size, void* d_ws, size_t ws_size,
                              hipStream_t stream)
{
    (void)in_sizes; (void)n_in; (void)out_size; (void)ws_size;
    const long B = NB, BK = NBK;

#define PF(i) ((const float*)d_in[i])
    const float* xi     = PF(0);
    const float* xj     = PF(1);
    const float* subemb = PF(2);
    const float* bnodes = PF(3);
    const float* bprior = PF(4);
    const float* bstats = PF(5);
    const float* bfeat  = PF(6);
    const unsigned char* bmask = (const unsigned char*)d_in[112];

    // bump allocator over d_ws
    char* wsp = (char*)d_ws;
    size_t off = 0;
    auto allocB = [&](size_t bytes) -> void* {
        void* p = (void*)(wsp + off);
        off += bytes;
        off = (off + 255) & ~(size_t)255;
        return p;
    };
    auto alloc  = [&](size_t elems) -> float* { return (float*)allocB(elems * 4); };
    auto allocH = [&](size_t elems) -> h16*   { return (h16*)allocB(elems * 2); };
    auto blks = [](long total) { return (unsigned)((total + 255) / 256); };

    // ---- weight transposes (padded, zero-filled, f16) ------------------
    struct WT { const h16* p; int kpad; };
    auto wt = [&](const float* W, int Kd, int N) -> WT {
        const int kpad = (Kd + 31) & ~31;
        const int npad = (N + 15) & ~15;
        h16* p = allocH((size_t)kpad * npad);
        k_wtrans16<<<blks((long)kpad * npad), 256, 0, stream>>>(W, p, Kd, N, kpad, npad);
        return WT{p, kpad};
    };

    const WT w_sub   = wt(PF(105), 128, 128);
    const WT w_mlp1  = wt(PF(65), 1152, 256);
    const WT w_mlp2  = wt(PF(67), 256, 2);
    const WT w_pair  = wt(PF(85), 1024, 256);
    const WT w_bq    = wt(PF(21), 1024, 128);
    const WT w_bn    = wt(PF(19), 256, 128);
    const WT w_bf    = wt(PF(13), 3, 128);
    const WT w_mtok  = wt(PF(71), 385, 256);       // kpad = 416
    const WT w_msc   = wt(PF(69), 256, 1);
    const WT w_mval  = wt(PF(75), 256, 256);
    const WT w_stat  = wt(PF(103), 4, 256);
    const WT w_rg1   = wt(PF(95), 512, 256),  w_rg2  = wt(PF(97), 256, 256);
    const WT w_rf1   = wt(PF(90), 512, 256),  w_rf2  = wt(PF(92), 256, 256);
    const WT w_sg1   = wt(PF(99), 512, 256),  w_sg2  = wt(PF(101), 256, 256);
    const WT w_c1    = wt(PF(25), 260, 128),  w_c2   = wt(PF(27), 128, 1);
    const WT w_lqw   = wt(PF(59), 1024, 256);
    const WT w_lc    = wt(PF(49), 128, 256);
    const WT w_lg1   = wt(PF(53), 516, 256),  w_lg2  = wt(PF(55), 256, 256);
    const WT w_loc   = wt(PF(61), 384, 256);
    const WT w_bmp   = wt(PF(15), 512, 256);
    const WT w_bo    = wt(PF(23), 256, 2);
    const WT w_cx    = wt(PF(29), 256, 256),  w_cs   = wt(PF(33), 256, 1);
    const WT w_g2l1  = wt(PF(37), 512, 256),  w_g2l2 = wt(PF(39), 256, 256);
    const WT w_p2lg1 = wt(PF(81), 512, 256),  w_p2lg2= wt(PF(83), 256, 256);
    const WT w_p2l1  = wt(PF(77), 512, 256),  w_p2l2 = wt(PF(79), 256, 256);
    const WT w_l2pg1 = wt(PF(45), 512, 256),  w_l2pg2= wt(PF(47), 256, 256);
    const WT w_l2p1  = wt(PF(41), 512, 256),  w_l2p2 = wt(PF(43), 256, 256);
    const WT w_bc1   = wt(PF(9), 517, 256),   w_bc2  = wt(PF(11), 256, 1);
    const WT w_do    = wt(PF(35), 256, 2);
    const WT w_u1    = wt(PF(109), 520, 256), w_u2   = wt(PF(111), 256, 1);
    const WT w_bankW = wt(PF(7), 16, 256);         // for la @ bank (kpad 32)
    h16* bankT16 = allocH(16 * 256);               // bank rows as Wt for lq @ bank^T
    k_cvt16<<<blks(16 * 256), 256, 0, stream>>>(PF(7), bankT16, 16 * 256);
    const WT w_bankT = WT{bankT16, 256};

    // ---- activations ---------------------------------------------------
    float* pair       = alloc(B * 1024);
    h16*   pair16     = allocH(B * 1024);
    h16*   pairN16    = allocH(B * 1024);
    h16*   subN16     = allocH(B * 128);
    float* psub       = alloc(B * 128);
    h16*   psubN16    = allocH(B * 128);
    h16*   cat115216  = allocH(B * 1152);
    h16*   h116       = allocH(B * 256);
    float* base_log   = alloc(B * 2);
    float* pair_state = alloc(B * 256);
    float* qb         = alloc(B * 128);
    h16*   bnodes16   = allocH(BK * 256);
    float* nodeH      = alloc(BK * 128);
    h16*   featIn16   = allocH(BK * 32);
    float* featH      = alloc(BK * 128);
    float* tokin      = alloc(BK * 416);
    h16*   tokin16    = allocH(BK * 416);
    h16*   tokens16   = allocH(BK * 256);
    float* scraw      = alloc(BK);
    float* attn       = alloc(BK);
    float* values     = alloc(BK * 256);
    float* expl       = alloc(B * 256);
    h16*   statsIn16  = allocH(B * 32);
    float* stats_rep  = alloc(B * 256);
    h16*   catb16     = allocH(B * CATS);
    h16*   t116       = allocH(B * 256);
    float* gbuf       = alloc(B * 256);
    float* mbuf       = alloc(B * 256);
    float* conf       = alloc(B);
    h16*   lq16       = allocH(B * 256);
    float* scoresM    = alloc(B * 16);
    float* laPf       = alloc(B * 32);
    h16*   laP16      = allocH(B * 32);
    float* labank     = alloc(B * 256);
    float* ctxl       = alloc(B * 256);
    float* lat        = alloc(B * 256);
    float* bls        = alloc(B * 256);
    float* blr        = alloc(B * 256);
    float* base_br    = alloc(B * 256);
    h16*   s16a       = allocH(B * 256);
    h16*   s16b       = allocH(B * 256);
    h16*   s16c       = allocH(B * 256);
    float* bbl        = alloc(B * 2);
    h16*   toks16     = allocH(B * 512);
    float* tproj      = alloc(B * 512);
    h16*   tproj16    = allocH(B * 512);
    float* tsc        = alloc(B * 2);
    float* lseed      = alloc(B * 256);
    float* local_comm = alloc(B * 256);
    float* staged     = alloc(B * 256);
    float* global_c   = alloc(B * 256);
    float* latres     = alloc(B * 256);
    float* bridge_rep = alloc(B * 256);
    float* bconf      = alloc(B);
    float* blogits    = alloc(B * 2);
    float* glogits    = alloc(B * 2);
    float* util       = alloc(B);
    float* outp       = (float*)d_out;

    auto gemm = [&](const h16* X, int ldx, WT w, const float* bias, float* Y,
                    h16* Y16, int ldy, long R, int N, int act) {
        launch_gemm(X, ldx, w.p, w.kpad, bias, Y, Y16, ldy, (int)R, N, act, stream);
    };
    auto copyh = [&](h16* dst, int ds, int dc, const float* src, int ss, int sc,
                     long rows, int cols) {
        k_copyh<<<blks(rows * (long)cols), 256, 0, stream>>>(dst, ds, dc, src, ss, sc, rows, cols);
    };
    auto zcolsh = [&](h16* dst, int stride, int col0, int cols, long rows) {
        k_fillcolsh<<<blks(rows * (long)cols), 256, 0, stream>>>(dst, stride, col0, cols, rows);
    };

    // ---- pair features, psub, base logits ------------------------------
    k_pairbuild<<<blks(B * 1024), 256, 0, stream>>>(xi, xj, pair, B, ND);
    k_cvt16<<<blks(B * 1024), 256, 0, stream>>>(pair, pair16, B * 1024);
    k_ln<float, h16><<<(unsigned)B, 256, 0, stream>>>(pair, pairN16, PF(87), PF(86), 1024, 1024);
    k_ln<float, h16><<<(unsigned)B, 256, 0, stream>>>(subemb, subN16, PF(107), PF(106), 128, 128);
    gemm(subN16, 128, w_sub, PF(104), psub, nullptr, 128, B, 128, ACT_RELU); // subgraph_lin
    copyh(cat115216, 1152, 0,    pair, 1024, 0, B, 1024);
    copyh(cat115216, 1152, 1024, psub, 128,  0, B, 128);
    gemm(cat115216, 1152, w_mlp1, PF(64), nullptr, h116, 256, B, 256, ACT_RELU); // mlp1
    gemm(h116, 256, w_mlp2, PF(66), base_log, nullptr, 2, B, 2, ACT_NONE);       // mlp2
    gemm(pairN16, 1024, w_pair, PF(84), pair_state, nullptr, 256, B, 256, ACT_RELU); // pair_lin

    // ---- explicit mediator state ---------------------------------------
    gemm(pair16, 1024, w_bq, PF(20), qb, nullptr, 128, B, 128, ACT_NONE);    // bquery
    k_cvt16<<<blks(BK * 256), 256, 0, stream>>>(bnodes, bnodes16, BK * 256);
    gemm(bnodes16, 256, w_bn, PF(18), nodeH, nullptr, 128, BK, 128, ACT_RELU); // bnode
    k_fillh<<<blks(BK * 32), 256, 0, stream>>>(featIn16, BK * 32);
    copyh(featIn16, 32, 0, bfeat, 3, 0, BK, 3);
    gemm(featIn16, 32, w_bf, PF(12), featH, nullptr, 128, BK, 128, ACT_RELU); // bfeat
    k_tokin<<<blks(BK * 385), 256, 0, stream>>>(nodeH, qb, featH, bprior, PF(88), tokin, BK);
    k_ln<float, h16><<<(unsigned)BK, 256, 0, stream>>>(tokin, tokin16, PF(73), PF(72), 416, 385);
    zcolsh(tokin16, 416, 385, 31, BK);
    gemm(tokin16, 416, w_mtok, PF(70), nullptr, tokens16, 256, BK, 256, ACT_RELU); // mtok_lin
    gemm(tokens16, 256, w_msc, PF(68), scraw, nullptr, 1, BK, 1, ACT_NONE);        // mscore
    k_softmax16<<<blks(B), 256, 0, stream>>>(scraw, bmask, attn, 16, B, 1.f);
    gemm(tokens16, 256, w_mval, PF(74), values, nullptr, 256, BK, 256, ACT_RELU);  // mvalue
    k_attnpool<<<blks(B * 256), 256, 0, stream>>>(attn, values, expl, B);
    k_fillh<<<blks(B * 32), 256, 0, stream>>>(statsIn16, B * 32);
    copyh(statsIn16, 32, 0, bstats, 4, 0, B, 4);
    gemm(statsIn16, 32, w_stat, PF(102), stats_rep, nullptr, 256, B, 256, ACT_RELU); // stats_proj

    // refine gate + mlp
    copyh(catb16, CATS, 0,   expl,      256, 0, B, 256);
    copyh(catb16, CATS, 256, stats_rep, 256, 0, B, 256);
    gemm(catb16, CATS, w_rg1, PF(94), nullptr, t116, 256, B, 256, ACT_RELU);  // rgate1
    gemm(t116, 256, w_rg2, PF(96), gbuf, nullptr, 256, B, 256, ACT_SIG);      // rgate2
    gemm(catb16, CATS, w_rf1, PF(89), nullptr, t116, 256, B, 256, ACT_RELU);  // refine1
    gemm(t116, 256, w_rf2, PF(91), mbuf, nullptr, 256, B, 256, ACT_NONE);     // refine2
    k_fma3<<<blks(B * 256), 256, 0, stream>>>(expl, gbuf, mbuf, expl, B * 256);
    // stats gate
    copyh(catb16, CATS, 0,   expl,      256, 0, B, 256);
    copyh(catb16, CATS, 256, stats_rep, 256, 0, B, 256);
    gemm(catb16, CATS, w_sg1, PF(98), nullptr, t116, 256, B, 256, ACT_RELU);  // sgate1
    gemm(t116, 256, w_sg2, PF(100), gbuf, nullptr, 256, B, 256, ACT_SIG);     // sgate2
    k_fma3<<<blks(B * 256), 256, 0, stream>>>(expl, gbuf, stats_rep, expl, B * 256);
    // conf
    copyh(catb16, CATS, 0,   expl,   256, 0, B, 256);
    copyh(catb16, CATS, 256, bstats, 4,   0, B, 4);
    zcolsh(catb16, CATS, 260, 28, B);                                         // pad to kpad=288
    gemm(catb16, CATS, w_c1, PF(24), nullptr, t116, 128, B, 128, ACT_RELU);   // conf1
    gemm(t116, 128, w_c2, PF(26), conf, nullptr, 1, B, 1, ACT_SIG);           // conf2

    // ---- latent state (lat is call-invariant) --------------------------
    gemm(pair16, 1024, w_lqw, PF(58), nullptr, lq16, 256, B, 256, ACT_NONE);  // latent_query
    gemm(lq16, 256, w_bankT, nullptr, scoresM, nullptr, 16, B, 16, ACT_NONE); // lq @ bank^T
    k_fill0<<<blks(B * 32), 256, 0, stream>>>(laPf, B * 32);
    k_softmax16<<<blks(B), 256, 0, stream>>>(scoresM, nullptr, laPf, 32, B, 1.f / 16.f);
    k_cvt16<<<blks(B * 32), 256, 0, stream>>>(laPf, laP16, B * 32);
    gemm(laP16, 32, w_bankW, nullptr, labank, nullptr, 256, B, 256, ACT_NONE); // la @ bank
    k_ln<float, h16><<<(unsigned)B, 256, 0, stream>>>(psub, psubN16, PF(51), PF(50), 128, 128);
    gemm(psubN16, 128, w_lc, PF(48), ctxl, nullptr, 256, B, 256, ACT_RELU);    // lat_ctx_lin
    k_addv<<<blks(B * 256), 256, 0, stream>>>(labank, ctxl, lat, B * 256);
    k_ln<float, float><<<(unsigned)B, 256, 0, stream>>>(lat, lat, PF(57), PF(56), 256, 256);

    auto latent_gate = [&](const float* gst, const float* lst, float* outg) {
        copyh(catb16, CATS, 0,   gst,    256, 0, B, 256);
        copyh(catb16, CATS, 256, lst,    256, 0, B, 256);
        copyh(catb16, CATS, 512, bstats, 4,   0, B, 4);
        zcolsh(catb16, CATS, 516, 28, B);                                     // pad to kpad=544
        gemm(catb16, CATS, w_lg1, PF(52), nullptr, t116, 256, B, 256, ACT_RELU);
        gemm(t116, 256, w_lg2, PF(54), outg, nullptr, 256, B, 256, ACT_SIG);
    };

    // ---- base bridge path ----------------------------------------------
    copyh(catb16, CATS, 0,   psub, 128, 0, B, 128);
    copyh(catb16, CATS, 128, expl, 256, 0, B, 256);
    k_ln<h16, h16><<<(unsigned)B, 256, 0, stream>>>(catb16, catb16, PF(63), PF(62), CATS, 384);
    gemm(catb16, CATS, w_loc, PF(60), bls, nullptr, 256, B, 256, ACT_RELU);   // local_lin
    latent_gate(pair_state, bls, gbuf);
    k_mul<<<blks(B * 256), 256, 0, stream>>>(gbuf, lat, blr, B * 256);
    copyh(catb16, CATS, 0,   bls,  256, 0, B, 256);
    copyh(catb16, CATS, 256, expl, 256, 0, B, 256);
    k_ln<h16, h16><<<(unsigned)B, 256, 0, stream>>>(catb16, catb16, PF(17), PF(16), CATS, 512);
    gemm(catb16, CATS, w_bmp, PF(14), base_br, nullptr, 256, B, 256, ACT_RELU); // bmp_lin
    k_addconf<<<blks(B * 256), 256, 0, stream>>>(base_br, conf, blr, 256, B * 256);
    k_cvt16<<<blks(B * 256), 256, 0, stream>>>(base_br, s16a, B * 256);
    gemm(s16a, 256, w_bo, PF(22), bbl, nullptr, 2, B, 2, ACT_NONE);           // bridge_out

    // ---- micro subgraph pooling ----------------------------------------
    copyh(toks16, 512, 0,   expl, 256, 0, B, 256);
    copyh(toks16, 512, 256, bls,  256, 0, B, 256);
    k_ln<h16, h16><<<(unsigned)(2 * B), 256, 0, stream>>>(toks16, toks16, PF(31), PF(30), 256, 256);
    gemm(toks16, 256, w_cx, PF(28), tproj, tproj16, 256, 2 * B, 256, ACT_RELU); // ctx_lin
    gemm(tproj16, 256, w_cs, PF(32), tsc, nullptr, 1, 2 * B, 1, ACT_NONE);      // ctx_score
    k_ctxpool<<<blks(B * 256), 256, 0, stream>>>(tsc, tproj, lseed, B);

    // g2l mixing
    copyh(catb16, CATS, 0,   pair_state, 256, 0, B, 256);
    copyh(catb16, CATS, 256, lseed,      256, 0, B, 256);
    gemm(catb16, CATS, w_g2l1, PF(36), nullptr, t116, 256, B, 256, ACT_RELU);
    gemm(t116, 256, w_g2l2, PF(38), gbuf, nullptr, 256, B, 256, ACT_SIG);
    k_mix<<<blks(B * 256), 256, 0, stream>>>(gbuf, pair_state, lseed, local_comm, B * 256);

    // pe = [pair_state, explicit]
    copyh(catb16, CATS, 0,   pair_state, 256, 0, B, 256);
    copyh(catb16, CATS, 256, expl,       256, 0, B, 256);
    gemm(catb16, CATS, w_p2lg1, PF(80), nullptr, t116, 256, B, 256, ACT_RELU);
    gemm(t116, 256, w_p2lg2, PF(82), gbuf, nullptr, 256, B, 256, ACT_SIG);
    gemm(catb16, CATS, w_p2l1, PF(76), nullptr, t116, 256, B, 256, ACT_RELU);
    gemm(t116, 256, w_p2l2, PF(78), mbuf, nullptr, 256, B, 256, ACT_NONE);
    k_fma3<<<blks(B * 256), 256, 0, stream>>>(lseed, gbuf, mbuf, staged, B * 256);
    k_avg<<<blks(B * 256), 256, 0, stream>>>(local_comm, staged, local_comm, B * 256);

    // ps = [pair_state, staged_local]
    copyh(catb16, CATS, 0,   pair_state, 256, 0, B, 256);
    copyh(catb16, CATS, 256, staged,     256, 0, B, 256);
    gemm(catb16, CATS, w_l2pg1, PF(44), nullptr, t116, 256, B, 256, ACT_RELU);
    gemm(t116, 256, w_l2pg2, PF(46), gbuf, nullptr, 256, B, 256, ACT_SIG);
    gemm(catb16, CATS, w_l2p1, PF(40), nullptr, t116, 256, B, 256, ACT_RELU);
    gemm(t116, 256, w_l2p2, PF(42), mbuf, nullptr, 256, B, 256, ACT_NONE);
    k_fma3<<<blks(B * 256), 256, 0, stream>>>(pair_state, gbuf, mbuf, global_c, B * 256);

    // ---- refined bridge path -------------------------------------------
    latent_gate(global_c, local_comm, gbuf);
    k_mul<<<blks(B * 256), 256, 0, stream>>>(gbuf, lat, latres, B * 256);
    copyh(catb16, CATS, 0,   local_comm, 256, 0, B, 256);
    copyh(catb16, CATS, 256, expl,       256, 0, B, 256);
    k_ln<h16, h16><<<(unsigned)B, 256, 0, stream>>>(catb16, catb16, PF(17), PF(16), CATS, 512);
    gemm(catb16, CATS, w_bmp, PF(14), bridge_rep, nullptr, 256, B, 256, ACT_RELU);
    k_addconf<<<blks(B * 256), 256, 0, stream>>>(bridge_rep, conf, latres, 256, B * 256);

    // bconf
    copyh(catb16, CATS, 0,   global_c,   256, 0, B, 256);
    copyh(catb16, CATS, 256, bridge_rep, 256, 0, B, 256);
    copyh(catb16, CATS, 512, bstats,     4,   0, B, 4);
    copyh(catb16, CATS, 516, conf,       1,   0, B, 1);
    zcolsh(catb16, CATS, 517, 27, B);                                         // pad to kpad=544
    gemm(catb16, CATS, w_bc1, PF(8), nullptr, t116, 256, B, 256, ACT_RELU);   // bconf1
    gemm(t116, 256, w_bc2, PF(10), bconf, nullptr, 1, B, 1, ACT_SIG);         // bconf2

    k_cvt16<<<blks(B * 256), 256, 0, stream>>>(bridge_rep, s16c, B * 256);
    k_cvt16<<<blks(B * 256), 256, 0, stream>>>(global_c, s16b, B * 256);
    gemm(s16c, 256, w_bo, PF(22), blogits, nullptr, 2, B, 2, ACT_NONE);       // bridge_out
    gemm(s16b, 256, w_do, PF(34), glogits, nullptr, 2, B, 2, ACT_NONE);       // direct_out

    // util
    copyh(catb16, CATS, 0,   global_c,   256, 0, B, 256);
    copyh(catb16, CATS, 256, bridge_rep, 256, 0, B, 256);
    copyh(catb16, CATS, 512, bstats,     4,   0, B, 4);
    copyh(catb16, CATS, 516, conf,       1,   0, B, 1);
    copyh(catb16, CATS, 517, bconf,      1,   0, B, 1);
    k_gap<<<blks(2 * B), 256, 0, stream>>>(glogits, blogits, catb16, CATS, 518, B);
    zcolsh(catb16, CATS, 520, 24, B);                                         // pad to kpad=544
    gemm(catb16, CATS, w_u1, PF(108), nullptr, t116, 256, B, 256, ACT_RELU);  // util1
    gemm(t116, 256, w_u2, PF(110), util, nullptr, 1, B, 1, ACT_SIG);          // util2

    k_final<<<blks(2 * B), 256, 0, stream>>>(glogits, blogits, bbl, base_log,
                                             bconf, util, PF(93), outp, B);
#undef PF
}
